// BaseGAT_Modified_37838661878217
// MI455X (gfx1250) — matrix operand order repaired and checked by reading, not compiled, not run on hardware
//
#include <hip/hip_runtime.h>
#include <hip/hip_bf16.h>

typedef __attribute__((ext_vector_type(16))) _Float16 v16h;
typedef __attribute__((ext_vector_type(8)))  float    v8f;

#define DN 64
#define DE 32
#define D_MSG 160

__device__ __forceinline__ void pack8(v16h& v, int base, float4 a, float4 b) {
    v[base + 0] = (_Float16)a.x; v[base + 1] = (_Float16)a.y;
    v[base + 2] = (_Float16)a.z; v[base + 3] = (_Float16)a.w;
    v[base + 4] = (_Float16)b.x; v[base + 5] = (_Float16)b.y;
    v[base + 6] = (_Float16)b.z; v[base + 7] = (_Float16)b.w;
}

// Kernel 1: per-edge attention score via WMMA.
// One wave computes scores for 16 consecutive edges:
//   A (16x32 f16 per chunk) = slice of z_edge = [h[src] | e | h[dst]]
//   B (32x16 f16)           = W slice replicated across all 16 columns
// 5 chunks of K=32 cover D_MSG=160. Accumulate f32, then leaky_relu and
// atomicMax (order-preserving uint key) into per-destination running max.
__global__ void edge_scores_wmma(const float* __restrict__ h,
                                 const float* __restrict__ e,
                                 const float* __restrict__ W,
                                 const int* __restrict__ src,
                                 const int* __restrict__ dst,
                                 float* __restrict__ a_out,
                                 unsigned int* __restrict__ m_key,
                                 int E) {
    const int lane = threadIdx.x & 31;
    const int wave = blockIdx.x * (blockDim.x >> 5) + (threadIdx.x >> 5);
    const int e0 = wave << 4;
    if (e0 >= E) return;

    const int m    = lane & 15;        // A-matrix row (edge within group)
    const int hi   = lane >> 4;        // half-wave id
    const int koff = hi << 3;          // K sub-offset: lanes 16-31 carry K+8
    const int edge = min(e0 + m, E - 1);

    const int   sN = src[edge];
    const int   dN = dst[edge];
    const float* zsrc = h + (size_t)sN * DN;
    const float* zdst = h + (size_t)dN * DN;
    const float* ze   = e + (size_t)edge * DE;

    v8f acc = {};
#pragma unroll
    for (int c = 0; c < 5; ++c) {
        // A layout (16-bit 16x32): lane carries K = koff..koff+7 (elems 0..7)
        // and K = 16+koff..16+koff+7 (elems 8..15) of chunk c.
        const int K1 = c * 32 + koff;
        const int K2 = K1 + 16;
        const float* p1;
        const float* p2;
        if (c < 2)       { p1 = zsrc + K1;        p2 = zsrc + K2;        }
        else if (c == 2) { p1 = ze   + (K1 - 64); p2 = ze   + (K2 - 64); }
        else             { p1 = zdst + (K1 - 96); p2 = zdst + (K2 - 96); }

        float4 f0 = *(const float4*)(p1);
        float4 f1 = *(const float4*)(p1 + 4);
        float4 f2 = *(const float4*)(p2);
        float4 f3 = *(const float4*)(p2 + 4);

        v16h afrag;
        pack8(afrag, 0, f0, f1);
        pack8(afrag, 8, f2, f3);

        // B layout (32x16): lane carries K = koff..koff+7 (elems 0..7) and
        // K = 16+koff..16+koff+7 (elems 8..15), matching A. Replicate W across all 16 columns so every
        // output column equals the desired dot product.
        const float* wp = W + c * 32 + (hi << 3);
        v16h bfrag;
#pragma unroll
        for (int i = 0; i < 16; ++i) bfrag[i] = (_Float16)wp[(i & 7) + ((i >> 3) << 4)];

        acc = __builtin_amdgcn_wmma_f32_16x16x32_f16(
            false, afrag, false, bfrag, (short)0, acc, false, false);
    }

    // D layout: lane 0 holds rows M=0..7 (edges e0..e0+7) in acc[0..7],
    // lane 16 holds rows M=8..15 (edges e0+8..e0+15).
    if (m == 0) {
#pragma unroll
        for (int r = 0; r < 8; ++r) {
            int eid = e0 + (hi << 3) + r;
            if (eid < E) {
                float v = acc[r];
                v = v > 0.0f ? v : 0.01f * v;          // leaky_relu(0.01)
                a_out[eid] = v;
                unsigned int bits = __float_as_uint(v);
                unsigned int key  = bits ^ ((bits & 0x80000000u) ? 0xFFFFFFFFu
                                                                 : 0x80000000u);
                atomicMax(&m_key[dst[eid]], key);      // global_atomic_max_u32
            }
        }
    }
}

// Kernel 2: ex = exp(a - max[dst]); segment-sum of ex per destination.
__global__ void edge_expsum(const float* __restrict__ a,
                            const int* __restrict__ dst,
                            const unsigned int* __restrict__ m_key,
                            float* __restrict__ ex,
                            float* __restrict__ s,
                            int E) {
    int i = blockIdx.x * blockDim.x + threadIdx.x;
    if (i >= E) return;
    int d = dst[i];
    unsigned int k    = m_key[d];
    unsigned int bits = (k & 0x80000000u) ? (k ^ 0x80000000u) : ~k;
    float mk = __uint_as_float(bits);
    float ev = __expf(a[i] - mk);
    ex[i] = ev;
    atomicAdd(&s[d], ev);                               // global_atomic_add_f32
}

// Kernel 3: the bandwidth-dominant scatter. One wave per edge; each lane
// contributes 5 of the 160 message floats via hardware f32 atomics into
// h_out[dst, et, :]. h gathers hit L2 (h = 12.8 MB << 192 MB L2).
__global__ void edge_scatter(const float* __restrict__ h,
                             const float* __restrict__ e,
                             const int* __restrict__ src,
                             const int* __restrict__ dst,
                             const int* __restrict__ et_idx,
                             const float* __restrict__ ex,
                             const float* __restrict__ s,
                             float* __restrict__ hout,
                             int E, int T) {
    const int lane = threadIdx.x & 31;
    const int eid  = blockIdx.x * (blockDim.x >> 5) + (threadIdx.x >> 5);
    if (eid >= E) return;

    const int d  = dst[eid];
    const int sN = src[eid];
    const int et = et_idx[eid];
    const float alpha = ex[eid] / s[d];

    const float* zsrc = h + (size_t)sN * DN;
    const float* zdst = h + (size_t)d  * DN;
    const float* ze   = e + (size_t)eid * DE;
    float* base = hout + ((size_t)d * T + et) * D_MSG;

    float v0 = zsrc[lane];
    float v1 = zsrc[lane + 32];
    float v2 = ze[lane];
    float v3 = zdst[lane];
    float v4 = zdst[lane + 32];
    atomicAdd(base + lane,       alpha * v0);
    atomicAdd(base + lane + 32,  alpha * v1);
    atomicAdd(base + lane + 64,  alpha * v2);
    atomicAdd(base + lane + 96,  alpha * v3);
    atomicAdd(base + lane + 128, alpha * v4);
}

extern "C" void kernel_launch(void* const* d_in, const int* in_sizes, int n_in,
                              void* d_out, int out_size, void* d_ws, size_t ws_size,
                              hipStream_t stream) {
    const float* h      = (const float*)d_in[0];
    const float* e      = (const float*)d_in[1];
    const float* W      = (const float*)d_in[2];
    const int*   src    = (const int*)d_in[3];
    const int*   dst    = (const int*)d_in[4];
    const int*   et_idx = (const int*)d_in[5];

    const int N = in_sizes[0] / DN;
    const int E = in_sizes[3];
    // recover n_types from output geometry: out = N*T*D_MSG + E*DE
    const int T = (int)(((size_t)out_size - (size_t)E * DE) /
                        ((size_t)N * D_MSG));

    float* h_out = (float*)d_out;
    float* e_out = h_out + (size_t)N * T * D_MSG;

    // workspace layout
    float*        a_sc  = (float*)d_ws;              // [E]
    float*        ex_sc = a_sc + E;                  // [E]
    unsigned int* m_key = (unsigned int*)(ex_sc + E);// [N]
    float*        s_sum = (float*)(m_key + N);       // [N]

    // zero h_out and the softmax state (m_key=0 is the minimal ordered key)
    hipMemsetAsync(h_out, 0, (size_t)N * T * D_MSG * sizeof(float), stream);
    hipMemsetAsync(m_key, 0, (size_t)N * 2 * sizeof(unsigned int), stream);

    // passthrough output: e
    hipMemcpyAsync(e_out, e, (size_t)E * DE * sizeof(float),
                   hipMemcpyDeviceToDevice, stream);

    // 1) scores (WMMA, wave per 16 edges; 8 waves per 256-thread block)
    {
        int waves  = (E + 15) / 16;
        int blocks = (waves + 7) / 8;
        edge_scores_wmma<<<blocks, 256, 0, stream>>>(h, e, W, src, dst,
                                                     a_sc, m_key, E);
    }
    // 2) exp + segment sum
    {
        int blocks = (E + 255) / 256;
        edge_expsum<<<blocks, 256, 0, stream>>>(a_sc, dst, m_key,
                                                ex_sc, s_sum, E);
    }
    // 3) normalized message scatter (wave per edge)
    {
        int blocks = (E + 7) / 8;
        edge_scatter<<<blocks, 256, 0, stream>>>(h, e, src, dst, et_idx,
                                                 ex_sc, s_sum, h_out, E, T);
    }
}
